// DeepAR_90374701843258
// MI455X (gfx1250) — compile-verified, matrix operand order checked by reading, not run on hardware
//
#include <hip/hip_runtime.h>
#include <hip/hip_bf16.h>
#include <math.h>

typedef __attribute__((ext_vector_type(16))) _Float16 v16h;
typedef __attribute__((ext_vector_type(8)))  float    v8f;

#define B_    32
#define LIN_  96
#define LOUT_ 24
#define N_    256
#define T1_   119          // LIN + LOUT - 1
#define H_    64
#define E_    32
#define G4_   256          // 4*H
#define ROWS_ 32           // sequences per workgroup
#define OSTRIDE_ (B_*T1_*N_)   // 974848 elements per output tensor

struct SM {
  _Float16 wih0[G4_*E_];      // [n][k], k=0..31  (B operand of emb GEMM)
  _Float16 whh0[G4_*H_];      // [n][k], k=0..63
  _Float16 wih1[G4_*H_];
  _Float16 whh1[G4_*H_];
  _Float16 h0[2][ROWS_*H_];   // double-buffered hidden state, fp16
  _Float16 h1[2][ROWS_*H_];
  _Float16 emb[ROWS_*E_];     // per-step embedded input (A operand)
  float bias0[G4_], bias1[G4_];   // bih+bhh folded
  float embW[E_], embB[E_];
  float muW[H_], sgW[H_];
  float meanS[ROWS_], stdS[ROWS_], invS[ROWS_];
  float mu_b, sg_b;
};

__device__ __forceinline__ float sigm(float x) { return 1.0f / (1.0f + __expf(-x)); }

// branch-free tanh: t = exp(-2|x|); tanh = sign(x) * (1-t)/(1+t)  (no EXEC splits)
__device__ __forceinline__ float ftanh(float x) {
  float t = __expf(-2.0f * fabsf(x));
  float r = (1.0f - t) / (1.0f + t);
  return copysignf(r, x);
}

// Load a 16x32 fp16 WMMA fragment (A or B) from LDS.
// base is [row][k] with leading dim ldk halves. Caller passes row = tileRow + (lane&15)
// and k = ktile + (lane>>4)*8 ; halves j0..7 -> k.., j8..15 -> k+16.. per ISA layout.
__device__ __forceinline__ v16h ld_frag(const _Float16* base, int row, int ldk, int k) {
  union { v16h v; uint4 q[2]; } f;
  const _Float16* p = base + row * ldk + k;
  f.q[0] = *reinterpret_cast<const uint4*>(p);
  f.q[1] = *reinterpret_cast<const uint4*>(p + 16);
  return f.v;
}

__device__ __forceinline__ v8f wmma16(v16h a, v16h b, v8f c) {
  return __builtin_amdgcn_wmma_f32_16x16x32_f16(false, a, false, b, (short)0, c, false, false);
}

extern __shared__ __align__(16) char smraw[];

__global__ __launch_bounds__(256) void deepar_lstm_kernel(
    const float* __restrict__ hd,  const float* __restrict__ fd,
    const float* __restrict__ hm,  const float* __restrict__ fm,
    const float* __restrict__ eps,
    const float* __restrict__ eW,  const float* __restrict__ eB,
    const float* __restrict__ Wih0, const float* __restrict__ Whh0,
    const float* __restrict__ bih0, const float* __restrict__ bhh0,
    const float* __restrict__ Wih1, const float* __restrict__ Whh1,
    const float* __restrict__ bih1, const float* __restrict__ bhh1,
    const float* __restrict__ muW,  const float* __restrict__ mub,
    const float* __restrict__ sgW,  const float* __restrict__ sgb,
    float* __restrict__ out)
{
  SM& sm = *reinterpret_cast<SM*>(smraw);
  const int tid = threadIdx.x;

  // ---------------- init: weights -> LDS fp16, biases folded, stats ----------------
  for (int i = tid; i < G4_*E_; i += 256) sm.wih0[i] = (_Float16)Wih0[i];
  for (int i = tid; i < G4_*H_; i += 256) sm.whh0[i] = (_Float16)Whh0[i];
  for (int i = tid; i < G4_*H_; i += 256) sm.wih1[i] = (_Float16)Wih1[i];
  for (int i = tid; i < G4_*H_; i += 256) sm.whh1[i] = (_Float16)Whh1[i];
  if (tid < G4_) { sm.bias0[tid] = bih0[tid] + bhh0[tid];
                   sm.bias1[tid] = bih1[tid] + bhh1[tid]; }
  if (tid < E_)  { sm.embW[tid] = eW[tid]; sm.embB[tid] = eB[tid]; }
  if (tid < H_)  { sm.muW[tid] = muW[tid]; sm.sgW[tid] = sgW[tid]; }
  if (tid == 0)  { sm.mu_b = mub[0]; sm.sg_b = sgb[0]; }
  for (int i = tid; i < ROWS_*H_; i += 256) {
    sm.h0[0][i] = (_Float16)0.0f; sm.h0[1][i] = (_Float16)0.0f;
    sm.h1[0][i] = (_Float16)0.0f; sm.h1[1][i] = (_Float16)0.0f;
  }
  if (tid < ROWS_) {
    int g = blockIdx.x * ROWS_ + tid;
    int b = g >> 8, n = g & 255;
    float s = 0.f, s2 = 0.f;
    for (int t = 0; t < LIN_; ++t) {
      float v = hd[(b*LIN_ + t)*N_ + n];
      s += v; s2 += v*v;
    }
    float mean = s * (1.0f/LIN_);
    float var  = s2 * (1.0f/LIN_) - mean*mean;
    float stdv = sqrtf(var + 1e-5f);
    sm.meanS[tid] = mean; sm.stdS[tid] = stdv; sm.invS[tid] = 1.0f / stdv;
  }
  __syncthreads();

  // ---------------- wave/tile mapping ----------------
  const int w    = tid >> 5;          // 8 waves
  const int lane = tid & 31;
  const int hs   = lane >> 4;         // half-wave select
  const int cl   = lane & 15;         // column within 16x16 tile
  const int rt   = w >> 2;            // row tile (0..1) -> rows rt*16..+16
  const int cg   = w & 3;             // H-column group (16 cols of H)
  const int rbase = rt * 16;
  const int kA   = hs * 8;            // fragment K base offset for this half-wave

  // step-invariant bias accumulator seeds (C operand of the first WMMA per chain)
  v8f bias0v[4], bias1v[4];
  #pragma unroll
  for (int q = 0; q < 4; ++q) {
    float b0 = sm.bias0[(q*4 + cg)*16 + cl];
    float b1 = sm.bias1[(q*4 + cg)*16 + cl];
    #pragma unroll
    for (int i = 0; i < 8; ++i) { bias0v[q][i] = b0; bias1v[q][i] = b1; }
  }

  // persistent cell state in registers (tile rows = rbase + v + 8*hs, col = cg*16+cl)
  float c0r[8], c1r[8];
  #pragma unroll
  for (int v = 0; v < 8; ++v) { c0r[v] = 0.f; c1r[v] = 0.f; }

  // per-thread row mapping for phase 1 (x + embedding): 8 threads per row
  const int r1 = tid >> 3;
  const int e0 = (tid & 7) * 4;
  const int g1 = blockIdx.x * ROWS_ + r1;
  const int b1 = g1 >> 8, n1 = g1 & 255;

  for (int t = 0; t < T1_; ++t) {
    _Float16* h0c = sm.h0[t & 1];
    _Float16* h0n = sm.h0[(t + 1) & 1];
    _Float16* h1c = sm.h1[t & 1];
    _Float16* h1n = sm.h1[(t + 1) & 1];

    // ---- phase 1: normalized input -> embedding (fp16) in LDS ----
    {
      float raw = (t < LIN_) ? hd[(b1*LIN_ + t)*N_ + n1]
                             : fd[(b1*LOUT_ + (t - LIN_))*N_ + n1];
      float x = (raw - sm.meanS[r1]) * sm.invS[r1];
      #pragma unroll
      for (int e = 0; e < 4; ++e)
        sm.emb[r1*E_ + e0 + e] = (_Float16)(x * sm.embW[e0 + e] + sm.embB[e0 + e]);
    }
    __syncthreads();   // emb ready; previous-step LDS reads all retired

    // ---- phase 2: layer-0 gates = bias + emb@Wih0^T + h0@Whh0^T, cell update ----
    {
      v16h aE  = ld_frag(sm.emb, rbase + cl, E_, kA);
      v16h aHa = ld_frag(h0c,    rbase + cl, H_, kA);
      v16h aHb = ld_frag(h0c,    rbase + cl, H_, 32 + kA);
      v8f acc[4];
      #pragma unroll
      for (int q = 0; q < 4; ++q) {
        const int nt = q*4 + cg;                 // gate q, 16-col slab cg
        v8f c = wmma16(aE,  ld_frag(sm.wih0, nt*16 + cl, E_, kA), bias0v[q]);
        c = wmma16(aHa, ld_frag(sm.whh0, nt*16 + cl, H_, kA),      c);
        c = wmma16(aHb, ld_frag(sm.whh0, nt*16 + cl, H_, 32 + kA), c);
        acc[q] = c;
      }
      #pragma unroll
      for (int v = 0; v < 8; ++v) {
        float ig = sigm(acc[0][v]), fg = sigm(acc[1][v]);
        float gg = ftanh(acc[2][v]), og = sigm(acc[3][v]);
        float cn = fg * c0r[v] + ig * gg;
        c0r[v] = cn;
        float hn = og * ftanh(cn);
        int m = rbase + v + 8*hs;
        h0n[m*H_ + cg*16 + cl] = (_Float16)hn;
      }
    }
    __syncthreads();   // h0_new visible

    // ---- phase 3: layer-1 gates = bias + h0_new@Wih1^T + h1@Whh1^T, cell update ----
    {
      v16h aXa = ld_frag(h0n, rbase + cl, H_, kA);
      v16h aXb = ld_frag(h0n, rbase + cl, H_, 32 + kA);
      v16h aHa = ld_frag(h1c, rbase + cl, H_, kA);
      v16h aHb = ld_frag(h1c, rbase + cl, H_, 32 + kA);
      v8f acc[4];
      #pragma unroll
      for (int q = 0; q < 4; ++q) {
        const int nt = q*4 + cg;
        v8f c = wmma16(aXa, ld_frag(sm.wih1, nt*16 + cl, H_, kA), bias1v[q]);
        c = wmma16(aXb, ld_frag(sm.wih1, nt*16 + cl, H_, 32 + kA), c);
        c = wmma16(aHa, ld_frag(sm.whh1, nt*16 + cl, H_, kA),      c);
        c = wmma16(aHb, ld_frag(sm.whh1, nt*16 + cl, H_, 32 + kA), c);
        acc[q] = c;
      }
      #pragma unroll
      for (int v = 0; v < 8; ++v) {
        float ig = sigm(acc[0][v]), fg = sigm(acc[1][v]);
        float gg = ftanh(acc[2][v]), og = sigm(acc[3][v]);
        float cn = fg * c1r[v] + ig * gg;
        c1r[v] = cn;
        float hn = og * ftanh(cn);
        int m = rbase + v + 8*hs;
        h1n[m*H_ + cg*16 + cl] = (_Float16)hn;
      }
    }
    __syncthreads();   // h1_new visible

    // ---- phase 4: Gaussian head + outputs (32 lanes; no WMMA below here) ----
    if (tid < ROWS_) {
      const int r = tid;
      const int g = blockIdx.x * ROWS_ + r;
      const int b = g >> 8, n = g & 255;
      float dm = 0.f, ds = 0.f;
      #pragma unroll 8
      for (int k = 0; k < H_; ++k) {
        float hv = (float)h1n[r*H_ + k];
        dm += hv * sm.muW[k];
        ds += hv * sm.sgW[k];
      }
      float mu   = dm + sm.mu_b;
      float sraw = ds + sm.sg_b;
      // branch-free stable softplus
      float sp   = fmaxf(sraw, 0.0f) + log1pf(__expf(-fabsf(sraw)));
      float sigma = sp + 1e-6f;

      float ev    = eps[(b*T1_ + t)*N_ + n];
      float mean  = sm.meanS[r], stdv = sm.stdS[r];
      float sample = mu + sigma * ev;
      float preds  = sample * stdv + mean;
      float mus    = mu    * stdv + mean;
      float sigmas = sigma * stdv + mean;   // matches reference (sigma*stdev + means)

      int tp1 = t + 1;
      float maskv = (tp1 < LIN_) ? hm[(b*LIN_ + tp1)*N_ + n]
                                 : fm[(b*LOUT_ + (tp1 - LIN_))*N_ + n];
      float realv = (tp1 < LIN_) ? hd[(b*LIN_ + tp1)*N_ + n]
                                 : fd[(b*LOUT_ + (tp1 - LIN_))*N_ + n];

      const int base = (b*T1_ + t)*N_ + n;
      out[0*OSTRIDE_ + base] = preds * maskv;
      out[1*OSTRIDE_ + base] = realv * maskv;
      out[2*OSTRIDE_ + base] = mus;
      out[3*OSTRIDE_ + base] = sigmas;
      out[4*OSTRIDE_ + base] = maskv;
    }
    // next iteration's phase-1 barrier orders head reads vs. future LDS writes
  }
}

extern "C" void kernel_launch(void* const* d_in, const int* in_sizes, int n_in,
                              void* d_out, int out_size, void* d_ws, size_t ws_size,
                              hipStream_t stream) {
  (void)in_sizes; (void)n_in; (void)out_size; (void)d_ws; (void)ws_size;
  const float* hd   = (const float*)d_in[0];
  const float* fd   = (const float*)d_in[1];
  const float* hm   = (const float*)d_in[2];
  const float* fm   = (const float*)d_in[3];
  const float* eps  = (const float*)d_in[4];
  const float* eW   = (const float*)d_in[5];
  const float* eB   = (const float*)d_in[6];
  const float* Wih0 = (const float*)d_in[7];
  const float* Whh0 = (const float*)d_in[8];
  const float* bih0 = (const float*)d_in[9];
  const float* bhh0 = (const float*)d_in[10];
  const float* Wih1 = (const float*)d_in[11];
  const float* Whh1 = (const float*)d_in[12];
  const float* bih1 = (const float*)d_in[13];
  const float* bhh1 = (const float*)d_in[14];
  const float* muW  = (const float*)d_in[15];
  const float* mub  = (const float*)d_in[16];
  const float* sgW  = (const float*)d_in[17];
  const float* sgb  = (const float*)d_in[18];
  float* out = (float*)d_out;

  const size_t smbytes = sizeof(SM);   // ~136 KB dynamic LDS (<< 320 KB/WGP)
  (void)hipFuncSetAttribute((const void*)deepar_lstm_kernel,
                            hipFuncAttributeMaxDynamicSharedMemorySize, (int)smbytes);

  dim3 grid((B_*N_) / ROWS_);          // 256 workgroups, each owns 32 sequences
  dim3 block(256);                     // 8 wave32
  deepar_lstm_kernel<<<grid, block, smbytes, stream>>>(
      hd, fd, hm, fm, eps, eW, eB,
      Wih0, Whh0, bih0, bhh0, Wih1, Whh1, bih1, bhh1,
      muW, mub, sgW, sgb, out);
}
